// GroupedQueryAttention_240518168980
// MI455X (gfx1250) — compile-verified, matrix operand order checked by reading, not run on hardware
//
#include <hip/hip_runtime.h>
#include <hip/hip_bf16.h>

// ---------------------------------------------------------------------------
// GQA block for MI455X (gfx1250, wave32, WMMA).
// f32->bf16 convert; QKV GEMM + bias + fused RoPE (V stored pre-transposed);
// flash attention with online softmax; out projection.  All matmuls via
// v_wmma_f32_16x16x32_bf16; global->LDS staging via global_load_async_to_lds
// (ASYNCcnt) with double-buffered LDS so DMA overlaps WMMA compute.
// Staging is straight-line (no exec masking), last chunk peeled.
// ---------------------------------------------------------------------------

typedef __attribute__((ext_vector_type(16))) __bf16 v16bf;
typedef __attribute__((ext_vector_type(8)))  float  v8f;

#define S_LEN 2048
#define DM    2048
#define DH    64
#define NKV   512          // KV_HEADS * DH
#define KTG   64           // GEMM K-chunk per LDS stage (2 WMMA k-steps)
#define LDSS  72           // LDS row stride (64 + pad, rows stay 16B-aligned)

__device__ __forceinline__ __bf16 to_bf16(float f) {
  unsigned u = __float_as_uint(f);
  u = (u + 0x7FFFu + ((u >> 16) & 1u)) >> 16;          // round-to-nearest-even
  unsigned short s = (unsigned short)u;
  return __builtin_bit_cast(__bf16, s);
}

// A/B fragment for v_wmma_f32_16x16x32_bf16: lane l holds row (l%16),
// k = (l/16)*8 + {0..7} (VGPRs 0-3) and 16 + (l/16)*8 + {0..7} (VGPRs 4-7).
// p = &tile[row*stride + step*32 + (l/16)*8]; 16B at p and 16B at p+16 elems.
__device__ __forceinline__ v16bf load_frag(const __bf16* p) {
  union { uint4 q[2]; v16bf v; } u;
  u.q[0] = *reinterpret_cast<const uint4*>(p);
  u.q[1] = *reinterpret_cast<const uint4*>(p + 16);
  return u.v;
}

__device__ __forceinline__ v8f wmma_bf16(v16bf a, v16bf b, v8f c) {
  return __builtin_amdgcn_wmma_f32_16x16x32_bf16(false, a, false, b,
                                                 (short)0, c, false, false);
}

// Async DMA: 16B per lane, global -> LDS, no VGPR round trip (ASYNCcnt).
// LDS byte offset = low 32 bits of the flat shared-memory address (ISA: LDS
// aperture truncates addr[31:0]).
__device__ __forceinline__ void async_load_b128(const __bf16* g, __bf16* l) {
  unsigned ldsoff = (unsigned)(unsigned long long)(const void*)l;
  asm volatile("global_load_async_to_lds_b128 %0, %1, off"
               :: "v"(ldsoff), "v"((unsigned long long)g)
               : "memory");
}

__device__ __forceinline__ void wait_async0() {
  asm volatile("s_wait_asynccnt 0" ::: "memory");
}

// ---------------------------------------------------------------------------
// Kernel 0: f32 -> bf16 conversion (4 elements / thread, b64 stores)
// ---------------------------------------------------------------------------
__global__ __launch_bounds__(256)
void cvt_kernel(const float* __restrict__ in, __bf16* __restrict__ out, int n) {
  int i = (blockIdx.x * 256 + threadIdx.x) * 4;
  if (i + 3 < n) {
    float4 v = *reinterpret_cast<const float4*>(in + i);
    union { __bf16 e[4]; unsigned long long u; } pk;
    pk.e[0] = to_bf16(v.x); pk.e[1] = to_bf16(v.y);
    pk.e[2] = to_bf16(v.z); pk.e[3] = to_bf16(v.w);
    *reinterpret_cast<unsigned long long*>(out + i) = pk.u;
  }
}

// ---------------------------------------------------------------------------
// Kernel 1: QKV projection GEMM + bias + fused RoPE epilogue.
//   N region [0,2048)    : Q -> rope -> Qo [S, 2048]
//   N region [2048,2560) : K -> rope -> Ko [S, 512]
//   N region [2560,3072) : V -> Vo stored TRANSPOSED [512, S]
// Grid (24,16): 128x128 tile / WG; 8 waves, each 32(M) x 64(N).
// Double-buffered async LDS staging, K-chunk = 64 (16 WMMAs per sync).
// ---------------------------------------------------------------------------
__global__ __launch_bounds__(256)
void qkv_rope_kernel(const __bf16* __restrict__ X,
                     const __bf16* __restrict__ Wq,
                     const __bf16* __restrict__ Wkv,
                     const float* __restrict__ bq,
                     const float* __restrict__ bkv,
                     __bf16* __restrict__ Qo,
                     __bf16* __restrict__ Ko,
                     __bf16* __restrict__ Vo) {
  __shared__ __align__(16) __bf16 As[2][128 * LDSS];
  __shared__ __align__(16) __bf16 Bs[2][128 * LDSS];

  const int tid = threadIdx.x, lane = tid & 31, wave = tid >> 5;
  const int lh = lane & 15, hi = lane >> 4;
  const int n0 = blockIdx.x * 128;
  const int m0 = blockIdx.y * 128;

  const __bf16* Wsrc;
  const float* bsrc;
  if (n0 < DM) { Wsrc = Wq  + (size_t)n0        * DM; bsrc = bq  + n0; }
  else         { Wsrc = Wkv + (size_t)(n0 - DM) * DM; bsrc = bkv + (n0 - DM); }
  const __bf16* Xb = X + (size_t)m0 * DM;

  const int wm = (wave >> 1) * 32;   // 4 waves along M
  const int wn = (wave & 1) * 64;    // 2 waves along N

  v8f acc[2][4];
  for (int i = 0; i < 2; i++)
    for (int j = 0; j < 4; j++)
      for (int e = 0; e < 8; e++) acc[i][j][e] = 0.f;

  // straight-line staging: thread owns (row = tid/8 + {0,32,64,96}, seg)
  const int srow = tid >> 3;
  const int sseg = (tid & 7) * 8;
  auto stage = [&](int buf, int kc) {
#pragma unroll
    for (int ro = 0; ro < 128; ro += 32) {
      async_load_b128(Xb   + (size_t)(srow + ro) * DM + kc + sseg,
                      &As[buf][(srow + ro) * LDSS + sseg]);
      async_load_b128(Wsrc + (size_t)(srow + ro) * DM + kc + sseg,
                      &Bs[buf][(srow + ro) * LDSS + sseg]);
    }
  };

  auto compute = [&](int buf) {
#pragma unroll
    for (int step = 0; step < 2; ++step) {
      const int koff = step * 32 + hi * 8;
      v16bf a0 = load_frag(&As[buf][(wm +  0 + lh) * LDSS + koff]);
      v16bf a1 = load_frag(&As[buf][(wm + 16 + lh) * LDSS + koff]);
      v16bf b0 = load_frag(&Bs[buf][(wn +  0 + lh) * LDSS + koff]);
      v16bf b1 = load_frag(&Bs[buf][(wn + 16 + lh) * LDSS + koff]);
      v16bf b2 = load_frag(&Bs[buf][(wn + 32 + lh) * LDSS + koff]);
      v16bf b3 = load_frag(&Bs[buf][(wn + 48 + lh) * LDSS + koff]);
      acc[0][0] = wmma_bf16(a0, b0, acc[0][0]);
      acc[0][1] = wmma_bf16(a0, b1, acc[0][1]);
      acc[0][2] = wmma_bf16(a0, b2, acc[0][2]);
      acc[0][3] = wmma_bf16(a0, b3, acc[0][3]);
      acc[1][0] = wmma_bf16(a1, b0, acc[1][0]);
      acc[1][1] = wmma_bf16(a1, b1, acc[1][1]);
      acc[1][2] = wmma_bf16(a1, b2, acc[1][2]);
      acc[1][3] = wmma_bf16(a1, b3, acc[1][3]);
    }
  };

  stage(0, 0);
  int buf = 0;
  for (int kc = 0; kc < DM - KTG; kc += KTG, buf ^= 1) {
    wait_async0();        // own async stores to LDS complete
    __syncthreads();      // everyone's complete; prev compute done
    stage(buf ^ 1, kc + KTG);
    compute(buf);
  }
  wait_async0();
  __syncthreads();
  compute(buf);           // peeled last chunk (no further staging)

  // epilogue: bias + RoPE (Q and K regions) + scatter store (V transposed)
  for (int i = 0; i < 2; i++) {
    for (int j = 0; j < 4; j++) {
      const int n = n0 + wn + j * 16 + lh;        // region uniform per tile
      const float bias = bsrc[wn + j * 16 + lh];
      for (int r = 0; r < 8; r++) {
        const int m = m0 + wm + i * 16 + r + hi * 8;
        float val = acc[i][j][r] + bias;
        if (n < DM + NKV) {  // RoPE for Q and K; interleaved pairs
          float other = __shfl_xor(val, 1, 32);   // partner (n^1) = lane^1
          int d = n & 63;
          float freq = __expf(-0.14391156f * (float)(d & ~1)); // 10000^(-2i/64)
          float sn, cs;
          __sincosf((float)m * freq, &sn, &cs);
          val = (d & 1) ? (other * sn + val * cs) : (val * cs - other * sn);
        }
        if (n < DM)
          Qo[(size_t)m * DM + n] = to_bf16(val);
        else if (n < DM + NKV)
          Ko[(size_t)m * NKV + (n - DM)] = to_bf16(val);
        else  // V pre-transposed: [kv_dim][S] so attention stages it directly
          Vo[(size_t)(n - DM - NKV) * S_LEN + m] = to_bf16(val);
      }
    }
  }
}

// ---------------------------------------------------------------------------
// Kernel 2: flash-style attention (no causal mask, per reference).
// Grid (16 qblocks, 32 heads); 8 waves, wave w owns 16 q rows.
// Per 64-key chunk (double-buffered async staging of K [key][d] and
// V^T [d][key]): S = Q K^T * 0.125, online softmax, P re-stripe via per-wave
// LDS, O += P V.
// ---------------------------------------------------------------------------
__global__ __launch_bounds__(256)
void attn_kernel(const __bf16* __restrict__ Qb, const __bf16* __restrict__ Kb,
                 const __bf16* __restrict__ Vt, __bf16* __restrict__ Ob) {
  __shared__ __align__(16) __bf16 Ks[2][64 * LDSS];     // [key][d]
  __shared__ __align__(16) __bf16 Vs[2][64 * LDSS];     // [d][key]
  __shared__ __align__(16) __bf16 Ps[8 * 16 * LDSS];    // per-wave P bounce

  const int tid = threadIdx.x, lane = tid & 31, wave = tid >> 5;
  const int lh = lane & 15, hi = lane >> 4;
  const int qb = blockIdx.x;            // 0..15
  const int h  = blockIdx.y;            // 0..31
  const int hk = h >> 2;                // GQA: 4 q-heads per kv-head
  const int qrow0 = qb * 128 + wave * 16;

  // Q A-fragments (head dim 64 => 2 k-steps), kept in registers
  v16bf qa[2];
#pragma unroll
  for (int s = 0; s < 2; s++)
    qa[s] = load_frag(Qb + (size_t)(qrow0 + lh) * DM + h * DH + s * 32 + hi * 8);

  float mrow[8], lrow[8];
  v8f o[4];
  for (int r = 0; r < 8; r++) { mrow[r] = -1e30f; lrow[r] = 0.f; }
  for (int t = 0; t < 4; t++)
    for (int e = 0; e < 8; e++) o[t][e] = 0.f;

  __bf16* Pw = &Ps[wave * 16 * LDSS];

  const int srow = tid >> 3;            // 0..31
  const int sseg = (tid & 7) * 8;       // 0..56
  auto stageKV = [&](int buf, int kc) {
    const __bf16* gk = Kb + (size_t)(kc + srow) * NKV + hk * DH + sseg;
    async_load_b128(gk,            &Ks[buf][srow * LDSS + sseg]);
    async_load_b128(gk + 32 * NKV, &Ks[buf][(srow + 32) * LDSS + sseg]);
    const __bf16* gv = Vt + (size_t)(hk * DH + srow) * S_LEN + kc + sseg;
    async_load_b128(gv,              &Vs[buf][srow * LDSS + sseg]);
    async_load_b128(gv + 32 * S_LEN, &Vs[buf][(srow + 32) * LDSS + sseg]);
  };

  auto process = [&](int buf) {
    // S tiles: 4 key sub-tiles of 16
    v8f sa[4];
    for (int j = 0; j < 4; j++)
      for (int e = 0; e < 8; e++) sa[j][e] = 0.f;
#pragma unroll
    for (int j = 0; j < 4; j++)
#pragma unroll
      for (int s = 0; s < 2; s++)
        sa[j] = wmma_bf16(
            qa[s], load_frag(&Ks[buf][(j * 16 + lh) * LDSS + s * 32 + hi * 8]),
            sa[j]);

    // online softmax per owned row r (rows r+8*hi live in this half-wave)
#pragma unroll
    for (int r = 0; r < 8; r++) {
      float mx = -1e30f;
      for (int j = 0; j < 4; j++) { sa[j][r] *= 0.125f; mx = fmaxf(mx, sa[j][r]); }
      for (int off = 1; off < 16; off <<= 1) mx = fmaxf(mx, __shfl_xor(mx, off, 32));
      float mnew  = fmaxf(mrow[r], mx);
      float alpha = __expf(mrow[r] - mnew);
      float rs = 0.f;
      for (int j = 0; j < 4; j++) {
        float p = __expf(sa[j][r] - mnew);
        sa[j][r] = p;
        rs += p;
      }
      for (int off = 1; off < 16; off <<= 1) rs += __shfl_xor(rs, off, 32);
      lrow[r] = lrow[r] * alpha + rs;
      mrow[r] = mnew;
      for (int t = 0; t < 4; t++) o[t][r] *= alpha;
    }

    // C-layout -> A-layout re-stripe of P through per-wave LDS
#pragma unroll
    for (int r = 0; r < 8; r++)
#pragma unroll
      for (int j = 0; j < 4; j++)
        Pw[(r + hi * 8) * LDSS + j * 16 + lh] = to_bf16(sa[j][r]);
    asm volatile("s_wait_dscnt 0" ::: "memory");  // per-wave LDS RAW fence

    // O += P V  (keys are the contraction dim: 2 k-steps of 32)
#pragma unroll
    for (int s = 0; s < 2; s++) {
      v16bf pa = load_frag(&Pw[lh * LDSS + s * 32 + hi * 8]);
#pragma unroll
      for (int t = 0; t < 4; t++)
        o[t] = wmma_bf16(
            pa, load_frag(&Vs[buf][(t * 16 + lh) * LDSS + s * 32 + hi * 8]),
            o[t]);
    }
  };

  stageKV(0, 0);
  int buf = 0;
  for (int kc = 0; kc < S_LEN - 64; kc += 64, buf ^= 1) {
    wait_async0();
    __syncthreads();
    stageKV(buf ^ 1, kc + 64);
    process(buf);
  }
  wait_async0();
  __syncthreads();
  process(buf);                         // peeled last key chunk

  // normalize and store attn output (bf16, [S, DM] heads concatenated)
  for (int t = 0; t < 4; t++)
    for (int r = 0; r < 8; r++) {
      const int m = qrow0 + r + hi * 8;
      Ob[(size_t)m * DM + h * DH + t * 16 + lh] = to_bf16(o[t][r] / lrow[r]);
    }
}

// ---------------------------------------------------------------------------
// Kernel 3: out projection  out = attn @ W_out.T + b_out  (f32 output)
// ---------------------------------------------------------------------------
__global__ __launch_bounds__(256)
void outproj_kernel(const __bf16* __restrict__ A, const __bf16* __restrict__ W,
                    const float* __restrict__ bias, float* __restrict__ out) {
  __shared__ __align__(16) __bf16 As[2][128 * LDSS];
  __shared__ __align__(16) __bf16 Bs[2][128 * LDSS];

  const int tid = threadIdx.x, lane = tid & 31, wave = tid >> 5;
  const int lh = lane & 15, hi = lane >> 4;
  const int n0 = blockIdx.x * 128;
  const int m0 = blockIdx.y * 128;
  const int wm = (wave >> 1) * 32;
  const int wn = (wave & 1) * 64;
  const __bf16* Ab = A + (size_t)m0 * DM;
  const __bf16* Wb = W + (size_t)n0 * DM;

  v8f acc[2][4];
  for (int i = 0; i < 2; i++)
    for (int j = 0; j < 4; j++)
      for (int e = 0; e < 8; e++) acc[i][j][e] = 0.f;

  const int srow = tid >> 3;
  const int sseg = (tid & 7) * 8;
  auto stage = [&](int buf, int kc) {
#pragma unroll
    for (int ro = 0; ro < 128; ro += 32) {
      async_load_b128(Ab + (size_t)(srow + ro) * DM + kc + sseg,
                      &As[buf][(srow + ro) * LDSS + sseg]);
      async_load_b128(Wb + (size_t)(srow + ro) * DM + kc + sseg,
                      &Bs[buf][(srow + ro) * LDSS + sseg]);
    }
  };

  auto compute = [&](int buf) {
#pragma unroll
    for (int step = 0; step < 2; ++step) {
      const int koff = step * 32 + hi * 8;
      v16bf a0 = load_frag(&As[buf][(wm +  0 + lh) * LDSS + koff]);
      v16bf a1 = load_frag(&As[buf][(wm + 16 + lh) * LDSS + koff]);
      v16bf b0 = load_frag(&Bs[buf][(wn +  0 + lh) * LDSS + koff]);
      v16bf b1 = load_frag(&Bs[buf][(wn + 16 + lh) * LDSS + koff]);
      v16bf b2 = load_frag(&Bs[buf][(wn + 32 + lh) * LDSS + koff]);
      v16bf b3 = load_frag(&Bs[buf][(wn + 48 + lh) * LDSS + koff]);
      acc[0][0] = wmma_bf16(a0, b0, acc[0][0]);
      acc[0][1] = wmma_bf16(a0, b1, acc[0][1]);
      acc[0][2] = wmma_bf16(a0, b2, acc[0][2]);
      acc[0][3] = wmma_bf16(a0, b3, acc[0][3]);
      acc[1][0] = wmma_bf16(a1, b0, acc[1][0]);
      acc[1][1] = wmma_bf16(a1, b1, acc[1][1]);
      acc[1][2] = wmma_bf16(a1, b2, acc[1][2]);
      acc[1][3] = wmma_bf16(a1, b3, acc[1][3]);
    }
  };

  stage(0, 0);
  int buf = 0;
  for (int kc = 0; kc < DM - KTG; kc += KTG, buf ^= 1) {
    wait_async0();
    __syncthreads();
    stage(buf ^ 1, kc + KTG);
    compute(buf);
  }
  wait_async0();
  __syncthreads();
  compute(buf);

  for (int i = 0; i < 2; i++)
    for (int j = 0; j < 4; j++) {
      const int n = n0 + wn + j * 16 + lh;
      const float b = bias[n];
      for (int r = 0; r < 8; r++) {
        const int m = m0 + wm + i * 16 + r + hi * 8;
        out[(size_t)m * DM + n] = acc[i][j][r] + b;
      }
    }
}

// ---------------------------------------------------------------------------
// Host-side launch
// ---------------------------------------------------------------------------
extern "C" void kernel_launch(void* const* d_in, const int* in_sizes, int n_in,
                              void* d_out, int out_size, void* d_ws, size_t ws_size,
                              hipStream_t stream) {
  const float* x     = (const float*)d_in[0];
  const float* W_q   = (const float*)d_in[1];
  const float* b_q   = (const float*)d_in[2];
  const float* W_kv  = (const float*)d_in[3];
  const float* b_kv  = (const float*)d_in[4];
  const float* W_out = (const float*)d_in[5];
  const float* b_out = (const float*)d_in[6];
  float* out = (float*)d_out;

  char* ws = (char*)d_ws;
  const size_t MB = 1024ull * 1024ull;
  __bf16* xb    = (__bf16*)(ws + 0 * MB);   // 8 MB  x bf16    [2048,2048]
  __bf16* wqb   = (__bf16*)(ws + 8 * MB);   // 8 MB  W_q bf16  [2048,2048]
  __bf16* wkvb  = (__bf16*)(ws + 16 * MB);  // 4 MB  W_kv bf16 [1024,2048]
  __bf16* woutb = (__bf16*)(ws + 20 * MB);  // 8 MB  W_out bf16
  __bf16* Qb    = (__bf16*)(ws + 28 * MB);  // 8 MB  roped Q   [2048,2048]
  __bf16* Kb    = (__bf16*)(ws + 36 * MB);  // 2 MB  roped K   [2048,512]
  __bf16* Vtb   = (__bf16*)(ws + 38 * MB);  // 2 MB  V^T       [512,2048]
  __bf16* Abuf  = (__bf16*)(ws + 40 * MB);  // 8 MB  attn out  [2048,2048]

  cvt_kernel<<<4194304 / 1024, 256, 0, stream>>>(x, xb, 4194304);
  cvt_kernel<<<4194304 / 1024, 256, 0, stream>>>(W_q, wqb, 4194304);
  cvt_kernel<<<2097152 / 1024, 256, 0, stream>>>(W_kv, wkvb, 2097152);
  cvt_kernel<<<4194304 / 1024, 256, 0, stream>>>(W_out, woutb, 4194304);

  qkv_rope_kernel<<<dim3(24, 16), 256, 0, stream>>>(xb, wqb, wkvb, b_q, b_kv,
                                                    Qb, Kb, Vtb);
  attn_kernel<<<dim3(16, 32), 256, 0, stream>>>(Qb, Kb, Vtb, Abuf);
  outproj_kernel<<<dim3(16, 16), 256, 0, stream>>>(Abuf, woutb, b_out, out);
}
